// SRUModel_20126216749610
// MI455X (gfx1250) — matrix operand order repaired and checked by reading, not compiled, not run on hardware
//
#include <hip/hip_runtime.h>
#include <hip/hip_bf16.h>
#include <math.h>

// ---------------------------------------------------------------------------
// SRU (2 layers, fused GEMM+scan) for MI455X / gfx1250, wave32 + WMMA bf16.
// T=2048, B=64, D=H=256, C=2.  Grid per layer: (HSPLIT=2, B=64) WGs x 256 thr.
// Weights register-resident as WMMA B-fragments; A tile double-buffered in
// LDS; layer-2 staging uses gfx1250 global_load_async_to_lds_b128 (ASYNCcnt);
// A-fragment k-loop manually software-pipelined (2 fragment buffers).
// ---------------------------------------------------------------------------

typedef __attribute__((ext_vector_type(16))) __bf16 v16bf;
typedef __attribute__((ext_vector_type(8)))  float  v8f;

#define T_TOT   2048
#define B_TOT   64
#define D_DIM   256
#define H_DIM   256
#define THREEH  768
#define HSPLIT  2
#define HCH     128          // channels per workgroup
#define TT      32           // T-tile rows (2 WMMA M-tiles)
#define NTILES  (T_TOT / TT) // 64 serial iterations
#define USTRIDE 388          // padded f32 stride for U tile (bank-conflict free)
#define ASTR32  132          // u32 (bf16-pair) stride for A row (256 cols + pad)
#define SCALE_X 1.41421356237309515f   // sqrt(1 + exp(2*HIGHWAY_BIAS)), bias=0

union FragB { v16bf v; unsigned u[8]; };
union FragF { v8f   v; float    f[8]; };

__device__ __forceinline__ unsigned short f32_bf16(float x) {
  union { float f; unsigned u; } c; c.f = x;
  unsigned r = c.u + 0x7FFFu + ((c.u >> 16) & 1u);   // round-to-nearest-even
  return (unsigned short)(r >> 16);
}
__device__ __forceinline__ float bf16_f32(unsigned short h) {
  union { unsigned u; float f; } c; c.u = ((unsigned)h) << 16;
  return c.f;
}
__device__ __forceinline__ float sigm(float x) { return 1.0f / (1.0f + __expf(-x)); }

// IN_BF16:  input tensor is bf16 [T,B,H] (layer 2) vs fp32 [T,B,D] (layer 1)
// LAST_ONLY: write only h[T-1] as fp32 [B,H] (layer 2) vs full bf16 [T,B,H]
template <bool IN_BF16, bool LAST_ONLY>
__global__ __launch_bounds__(256) void sru_layer_kernel(
    const void* __restrict__ in,
    const float* __restrict__ W,      // [256, 768] fp32
    const float* __restrict__ bias,   // [512] fp32  (b_f | b_r)
    void* __restrict__ out)
{
  const int hb   = blockIdx.x;        // channel-half: 0..1
  const int b    = blockIdx.y;        // batch: 0..63
  const int tid  = threadIdx.x;
  const int wave = tid >> 5;
  const int lane = tid & 31;

  __shared__ unsigned ldsA[2][TT * ASTR32]; // A tiles, bf16 pairs (32 x 256)
  __shared__ float    ldsU[TT * USTRIDE];   // U tile, f32 (32 x 384, padded)

  // ---- one-time: preload this wave's W slice as WMMA B fragments (bf16) ----
  // B (32x16 bf16) layout: N = lane%16; VGPR v holds K = 8h+2v,+1 (v<4) / 16+8h+2(v-4),+1 (v>=4), h = lane/16.
  FragB Breg[3][8];
  {
    #pragma unroll
    for (int nt = 0; nt < 3; ++nt) {
      const int j    = wave * 3 + nt;                              // n-tile 0..23
      const int col0 = (j >> 3) * H_DIM + hb * HCH + (j & 7) * 16; // global col
      #pragma unroll
      for (int ks = 0; ks < 8; ++ks) {
        const int kb = (lane >> 4) << 3;
        const float* wr = W + col0 + (lane & 15);
        #pragma unroll
        for (int v = 0; v < 8; ++v) {
          const int k = ks * 32 + kb + (v < 4 ? 2 * v : 16 + 2 * (v - 4));
          unsigned short lo = f32_bf16(wr[(long)k * THREEH]);
          unsigned short hi = f32_bf16(wr[(long)(k + 1) * THREEH]);
          Breg[nt][ks].u[v] = (unsigned)lo | ((unsigned)hi << 16);
        }
      }
    }
  }

  // ---- per-channel scan state (threads 0..127 own one channel each) ----
  const int chg = hb * HCH + tid;      // global channel (valid when tid<128)
  float c_state = 0.0f, b_f = 0.0f, b_r = 0.0f;
  if (tid < HCH) { b_f = bias[chg]; b_r = bias[H_DIM + chg]; }

  const float*    inF = (const float*)in;
  unsigned short* outH    = (unsigned short*)out;
  float*          outLast = (float*)out;

  // ---- staging helpers ----
  auto stage_f32 = [&](int t0n, int buf) {   // global fp32 -> bf16 pairs in LDS
    #pragma unroll
    for (int i = 0; i < (TT * 128) / 256; ++i) {    // 16 pair-slots / thread
      const int u   = i * 256 + tid;
      const int row = u >> 7;
      const int kk  = u & 127;
      const float* p = inF + ((long)(t0n + row) * B_TOT + b) * D_DIM + 2 * kk;
      unsigned pack = (unsigned)f32_bf16(p[0]) | ((unsigned)f32_bf16(p[1]) << 16);
      ldsA[buf][row * ASTR32 + kk] = pack;
    }
  };
  auto stage_bf16_async = [&](int t0n, int buf) {  // async copy, no VGPR round-trip
    const unsigned base = (unsigned)(size_t)&ldsA[buf][0];
    #pragma unroll
    for (int i = 0; i < (TT * 512) / (256 * 16); ++i) {   // 4 x 16B / thread
      const int chunk = i * 256 + tid;
      const int row   = chunk >> 5;       // 32 x 16B chunks per row
      const int c16   = chunk & 31;
      const char* g = (const char*)in +
          (((long)(t0n + row) * B_TOT + b) * (H_DIM * 2)) + c16 * 16;
      const unsigned loff = base + row * (ASTR32 * 4) + c16 * 16;
      asm volatile("global_load_async_to_lds_b128 %0, %1, off"
                   :: "v"(loff), "v"(g) : "memory");
    }
  };
  // load both M-tiles' A fragments for one k-step
  auto load_afrag = [&](FragB (&a)[2], int cur, int ks) {
    #pragma unroll
    for (int mt = 0; mt < 2; ++mt) {
      const int m  = mt * 16 + (lane & 15);
      const int kb = (lane >> 4) << 3;              // 0 or 8
      #pragma unroll
      for (int v = 0; v < 8; ++v) {
        const int k = ks * 32 + kb + (v < 4 ? 2 * v : 16 + 2 * (v - 4));
        a[mt].u[v] = ldsA[cur][m * ASTR32 + (k >> 1)];
      }
    }
  };

  // ---- prologue: stage tile 0 into buffer 0 ----
  if constexpr (IN_BF16) {
    stage_bf16_async(0, 0);
    asm volatile("s_wait_asynccnt 0x0" ::: "memory");
  } else {
    stage_f32(0, 0);
  }

  for (int ti = 0; ti < NTILES; ++ti) {
    const int t0  = ti * TT;
    const int cur = ti & 1;
    const int nxt = cur ^ 1;
    __syncthreads();   // publish ldsA[cur]; previous scan done with ldsU

    // kick off next tile's async staging immediately (overlaps GEMM + scan)
    if constexpr (IN_BF16) {
      if (ti + 1 < NTILES) stage_bf16_async(t0 + TT, nxt);
    }
    if (ti + 2 < NTILES) {  // L2 prefetch hint two tiles ahead
      const long nb = (long)(t0 + 2 * TT) * B_TOT + b;
      __builtin_prefetch(IN_BF16
          ? (const void*)((const char*)in + nb * (H_DIM * 2))
          : (const void*)(inF + nb * D_DIM), 0, 0);
    }

    // ---- GEMM: U(32x384) = A(32x256) * Wslice(256x384) via WMMA bf16 ----
    // k-loop software-pipelined: load frags for ks+1 before WMMAs of ks.
    FragF acc[2][3];
    #pragma unroll
    for (int mt = 0; mt < 2; ++mt)
      #pragma unroll
      for (int nt = 0; nt < 3; ++nt)
        #pragma unroll
        for (int e = 0; e < 8; ++e) acc[mt][nt].f[e] = 0.0f;

    FragB aP[2], aQ[2];                 // ping-pong fragment buffers
    load_afrag(aP, cur, 0);
    #pragma unroll
    for (int ks = 0; ks < 8; ++ks) {
      FragB (&aC)[2] = (ks & 1) ? aQ : aP;   // current (already loaded)
      FragB (&aN)[2] = (ks & 1) ? aP : aQ;   // next (load now, use next iter)
      if (ks + 1 < 8) load_afrag(aN, cur, ks + 1);
      #pragma unroll
      for (int mt = 0; mt < 2; ++mt)
        #pragma unroll
        for (int nt = 0; nt < 3; ++nt)
          acc[mt][nt].v = __builtin_amdgcn_wmma_f32_16x16x32_bf16(
              false, aC[mt].v, false, Breg[nt][ks].v,
              (short)0, acc[mt][nt].v, false, false);
    }

    // layer-1 staging after WMMA issue: global load latency overlaps WMMAs
    if constexpr (!IN_BF16) {
      if (ti + 1 < NTILES) stage_f32(t0 + TT, nxt);
    }

    // ---- spill accumulators to U tile (C layout: VGPR r -> M=r / M=8+r) ----
    {
      const int m0 = (lane >> 4) << 3;            // 0 or 8
      const int nl = lane & 15;
      #pragma unroll
      for (int mt = 0; mt < 2; ++mt)
        #pragma unroll
        for (int nt = 0; nt < 3; ++nt) {
          const int j    = wave * 3 + nt;
          const int ucol = (j >> 3) * HCH + (j & 7) * 16 + nl;  // local 0..383
          #pragma unroll
          for (int r = 0; r < 8; ++r)
            ldsU[(mt * 16 + m0 + r) * USTRIDE + ucol] = acc[mt][nt].f[r];
        }
    }
    __syncthreads();   // U visible; ldsA[cur] still intact for skip reads

    // ---- sequential SRU scan over the 32 rows (one channel per thread) ----
    if (tid < HCH) {
      const int cl = tid;
      for (int row = 0; row < TT; ++row) {
        const float xt = ldsU[row * USTRIDE + cl];
        const float fr = ldsU[row * USTRIDE + HCH + cl];
        const float rr = ldsU[row * USTRIDE + 2 * HCH + cl];
        const float f  = sigm(fr + b_f);          // HIGHWAY_BIAS == 0
        c_state = f * c_state + (1.0f - f) * xt;
        const float r  = sigm(rr + b_r);
        const unsigned pk = ldsA[cur][row * ASTR32 + (chg >> 1)];
        const float skip = bf16_f32((unsigned short)((chg & 1) ? (pk >> 16)
                                                               : (pk & 0xFFFFu)));
        const float h = r * c_state + (1.0f - r) * skip * SCALE_X;
        if (LAST_ONLY) {
          if (t0 + row == T_TOT - 1) outLast[b * H_DIM + chg] = h;
        } else {
          outH[((long)(t0 + row) * B_TOT + b) * H_DIM + chg] = f32_bf16(h);
        }
      }
    }

    // async writes into ldsA[nxt] must land before next top-of-loop barrier
    if constexpr (IN_BF16) {
      asm volatile("s_wait_asynccnt 0x0" ::: "memory");
    }
  }
}

// ---- head: out[b,c] = sum_h h2last[b,h] * lin_W[h,c] + lin_b[c] ----
__global__ __launch_bounds__(256) void sru_head_kernel(
    const float* __restrict__ hlast,   // [B, H] fp32
    const float* __restrict__ linW,    // [H, 2] fp32
    const float* __restrict__ linb,    // [2]
    float* __restrict__ out)           // [B, 2]
{
  __shared__ float red0[256];
  __shared__ float red1[256];
  const int b = blockIdx.x, tid = threadIdx.x;
  const float hv = hlast[b * H_DIM + tid];
  red0[tid] = hv * linW[tid * 2 + 0];
  red1[tid] = hv * linW[tid * 2 + 1];
  __syncthreads();
  for (int s = 128; s > 0; s >>= 1) {
    if (tid < s) { red0[tid] += red0[tid + s]; red1[tid] += red1[tid + s]; }
    __syncthreads();
  }
  if (tid == 0) {
    out[b * 2 + 0] = red0[0] + linb[0];
    out[b * 2 + 1] = red1[0] + linb[1];
  }
}

extern "C" void kernel_launch(void* const* d_in, const int* in_sizes, int n_in,
                              void* d_out, int out_size, void* d_ws, size_t ws_size,
                              hipStream_t stream) {
  (void)in_sizes; (void)n_in; (void)out_size; (void)ws_size;
  const float* x  = (const float*)d_in[0];   // [T,B,D]
  const float* W1 = (const float*)d_in[1];   // [D,3H]
  const float* b1 = (const float*)d_in[2];   // [2H]
  const float* W2 = (const float*)d_in[3];   // [H,3H]
  const float* b2 = (const float*)d_in[4];   // [2H]
  const float* lW = (const float*)d_in[5];   // [H,C]
  const float* lb = (const float*)d_in[6];   // [C]
  float* out = (float*)d_out;                // [B,C]

  unsigned short* h1 = (unsigned short*)d_ws;                        // bf16 [T,B,H]
  float* h2last = (float*)((char*)d_ws +
                           (size_t)T_TOT * B_TOT * H_DIM * sizeof(unsigned short));

  dim3 grid(HSPLIT, B_TOT), blk(256);
  sru_layer_kernel<false, false><<<grid, blk, 0, stream>>>(x,  W1, b1, h1);
  sru_layer_kernel<true,  true ><<<grid, blk, 0, stream>>>(h1, W2, b2, h2last);
  sru_head_kernel<<<dim3(B_TOT), dim3(256), 0, stream>>>(h2last, lW, lb, out);
}
